// FDCN_90486370992652
// MI455X (gfx1250) — matrix operand order repaired and checked by reading, not compiled, not run on hardware
//
#include <hip/hip_runtime.h>
#include <hip/hip_bf16.h>
#include <math.h>

#define DEVFN __device__ __forceinline__

typedef __attribute__((ext_vector_type(16))) _Float16 v16h;
typedef __attribute__((ext_vector_type(8)))  _Float16 v8h;
typedef __attribute__((ext_vector_type(8)))  float    v8f;

// ---- fixed problem dims (from reference setup_inputs) ----
constexpr int Nn  = 4;
constexpr int Cc  = 64;
constexpr int NET = 32;
constexpr int Hh  = 128;
constexpr int Ww  = 128;
constexpr int Gg  = 4;
constexpr int GC  = 16;
constexpr int HW  = Hh * Ww;
constexpr int NPIX = Nn * HW;              // 65536 pixels
constexpr float BN_EPS = 1e-5f;
constexpr float LN_EPS = 1e-6f;

// ---- WMMA helpers (CDNA5 16x16x32 f16 -> f32) ----
DEVFN v8f wmma_f16(v16h a, v16h b, v8f c) {
  return __builtin_amdgcn_wmma_f32_16x16x32_f16(false, a, false, b, (short)0, c, false, false);
}
// A-matrix 16x32 f16 layout (ISA 7.12.2): lane = M (mod 16);
// lanes 0-15 hold K = {0..7, 16..23}, lanes 16-31 hold K = {8..15, 24..31}
// -> each half of the fragment is 8 *consecutive* K values (one b128 from NHWC f16)
DEVFN int bK(int lane, int e) { return (e & 7) + ((lane & 16) ? 8 : 0) + ((e & 8) ? 16 : 0); }

DEVFN float sigm(float x) { return 1.0f / (1.0f + __expf(-x)); }

DEVFN v16h combine(v8h lo, v8h hi) {
  v16h a;
#pragma unroll
  for (int e = 0; e < 8; ++e) { a[e] = lo[e]; a[e + 8] = hi[e]; }
  return a;
}

// A tile from channels-last f16 buffer [NPIX x 64]: two global_load_b128 per lane
DEVFN v16h gA_nhwc_h(const _Float16* __restrict__ src, int pbase, int kc, int lane) {
  const _Float16* row = src + (size_t)(pbase + (lane & 15)) * 64;
  int base = kc * 32 + ((lane & 16) ? 8 : 0);
  v8h lo = *(const v8h*)(row + base);
  v8h hi = *(const v8h*)(row + base + 16);
  return combine(lo, hi);
}
// implicit-GEMM A gather for 3x3 convs: K = t*64 + c. Each 8-run has a fixed tap t,
// so it is one bounds check + one b128 per half-fragment.
DEVFN v16h gA_gru(const _Float16* __restrict__ src64, int pbase, int kc, int lane) {
  int p = pbase + (lane & 15);
  int n = p / HW, hw = p % HW, h = hw / Ww, w = hw % Ww;
  int base = (lane & 16) ? 8 : 0;
  v8h parts[2];
#pragma unroll
  for (int r = 0; r < 2; ++r) {
    int k0 = kc * 32 + base + r * 16;         // multiple of 8; run stays in one tap
    int t = k0 >> 6, c0 = k0 & 63;
    int hh = h + t / 3 - 1, ww2 = w + t % 3 - 1;
    v8h v = {};
    if (hh >= 0 && hh < Hh && ww2 >= 0 && ww2 < Ww)
      v = *(const v8h*)(src64 + ((size_t)((n * Hh + hh) * Ww + ww2)) * 64 + c0);
    parts[r] = v;
  }
  return combine(parts[0], parts[1]);
}

// ---- LDS weight staging (pre-swizzled into WMMA B layout) ----
// slot layout: (((kc*nnt + nt)*32 + lane)*16 + e), 32 B per lane-slot -> ds_load_b128 x2
DEVFN v16h gB_lds(const _Float16* lds, int nnt, int kc, int nt, int lane) {
  return *(const v16h*)(lds + ((size_t)((kc * nnt + nt) * 32 + lane) << 4));
}
DEVFN void stage_dense(_Float16* lds, const float* __restrict__ w,
                       int Nout, int Ktot, int nkc, int nnt) {
  int tot = nkc * nnt * 512;                  // always a multiple of 256
  for (int idx = threadIdx.x; idx < tot; idx += 256) {
    int e = idx & 15, lane = (idx >> 4) & 31;
    int nt = (idx >> 9) % nnt, kc = idx / (512 * nnt);
    int k = kc * 32 + bK(lane, e);
    int o = nt * 16 + (lane & 15);
    float v = (o < Nout && k < Ktot) ? w[(size_t)o * Ktot + k] : 0.0f;
    lds[idx] = (_Float16)v;
  }
}
// GRU conv weight (32, 64, 3, 3); K = t*64 + c
DEVFN void stage_gru(_Float16* lds, const float* __restrict__ w) {
  const int nnt = 2, tot = 18 * nnt * 512;
  for (int idx = threadIdx.x; idx < tot; idx += 256) {
    int e = idx & 15, lane = (idx >> 4) & 31;
    int nt = (idx >> 9) % nnt, kc = idx / (512 * nnt);
    int k = kc * 32 + bK(lane, e);
    int t = k >> 6, c = k & 63;
    int o = nt * 16 + (lane & 15);
    lds[idx] = (_Float16)w[((size_t)(o * 64 + c)) * 9 + t];
  }
}

// =====================================================================
// 1) pack input_x -> xh (f16 NHWC64) and input_y -> hx channels [0,32)
__global__ void k_pack_xy(const float* __restrict__ x, const float* __restrict__ y,
                          _Float16* __restrict__ xh, _Float16* __restrict__ hx) {
  int i = blockIdx.x * blockDim.x + threadIdx.x;
  if (i >= NPIX * 64) return;
  int p = i >> 6, c = i & 63;
  int n = p / HW, hw = p % HW;
  xh[(size_t)p * 64 + c] = (_Float16)x[((size_t)(n * Cc + c)) * HW + hw];
  if (c < NET)
    hx[(size_t)p * 64 + c] = (_Float16)y[((size_t)(n * NET + c)) * HW + hw];
}

// 2) encoder: 1x1 conv 64->32 + BN(eval) + SiLU -> hx channels [32,64)
__global__ void k_encoder(const _Float16* __restrict__ xh, const float* __restrict__ ew,
                          const float* __restrict__ eb, const float* __restrict__ bg,
                          const float* __restrict__ bb, const float* __restrict__ bm,
                          const float* __restrict__ bv, _Float16* __restrict__ hx) {
  __shared__ __align__(32) _Float16 wL[2 * 2 * 512];
  stage_dense(wL, ew, 32, 64, 2, 2);
  __syncthreads();
  int lane = threadIdx.x & 31, wave = threadIdx.x >> 5;
  int pbase = (blockIdx.x * 8 + wave) * 16;
  v8f acc[2] = {};
#pragma unroll
  for (int kc = 0; kc < 2; ++kc) {
    v16h a = gA_nhwc_h(xh, pbase, kc, lane);
#pragma unroll
    for (int nt = 0; nt < 2; ++nt) acc[nt] = wmma_f16(a, gB_lds(wL, 2, kc, nt, lane), acc[nt]);
  }
#pragma unroll
  for (int nt = 0; nt < 2; ++nt) {
    int o = nt * 16 + (lane & 15);
    float sc = bg[o] * rsqrtf(bv[o] + BN_EPS);
    float sh = bb[o] - bm[o] * sc;
    float bo = eb[o];
#pragma unroll
    for (int v = 0; v < 8; ++v) {
      int m = v + ((lane & 16) ? 8 : 0);
      float e = (acc[nt][v] + bo) * sc + sh;
      hx[(size_t)(pbase + m) * 64 + 32 + o] = (_Float16)(e * sigm(e));
    }
  }
}

// 3) GRU z,r: two 3x3 convs (implicit GEMM, K=576) sharing the A gather
__global__ void k_gru_zr(const _Float16* __restrict__ hx,
                         const float* __restrict__ wz, const float* __restrict__ bz,
                         const float* __restrict__ wr, const float* __restrict__ br,
                         float* __restrict__ zb, float* __restrict__ rb) {
  __shared__ __align__(32) _Float16 wzL[18 * 2 * 512];
  __shared__ __align__(32) _Float16 wrL[18 * 2 * 512];
  __builtin_prefetch(wz, 0, 3);
  __builtin_prefetch(wr, 0, 3);
  stage_gru(wzL, wz);
  stage_gru(wrL, wr);
  __syncthreads();
  int lane = threadIdx.x & 31, wave = threadIdx.x >> 5;
  int pbase = (blockIdx.x * 8 + wave) * 16;
  v8f az[2] = {}, ar[2] = {};
#pragma unroll 1
  for (int kc = 0; kc < 18; ++kc) {
    v16h a = gA_gru(hx, pbase, kc, lane);
#pragma unroll
    for (int nt = 0; nt < 2; ++nt) {
      az[nt] = wmma_f16(a, gB_lds(wzL, 2, kc, nt, lane), az[nt]);
      ar[nt] = wmma_f16(a, gB_lds(wrL, 2, kc, nt, lane), ar[nt]);
    }
  }
#pragma unroll
  for (int nt = 0; nt < 2; ++nt) {
    int o = nt * 16 + (lane & 15);
    float z0 = bz[o], r0 = br[o];
#pragma unroll
    for (int v = 0; v < 8; ++v) {
      int m = v + ((lane & 16) ? 8 : 0);
      size_t idx = (size_t)(pbase + m) * 32 + o;
      zb[idx] = sigm(az[nt][v] + z0);
      rb[idx] = sigm(ar[nt][v] + r0);
    }
  }
}

// 4) build a2 = concat([r*y, x_en]) in f16 NHWC(64)
__global__ void k_build_a2(const float* __restrict__ y, const float* __restrict__ rb,
                           const _Float16* __restrict__ hx, _Float16* __restrict__ a2) {
  int i = blockIdx.x * blockDim.x + threadIdx.x;
  if (i >= NPIX * 64) return;
  int p = i >> 6, c = i & 63;
  _Float16 v;
  if (c < 32) {
    int n = p / HW, hw = p % HW;
    v = (_Float16)(rb[(size_t)p * 32 + c] * y[((size_t)(n * NET + c)) * HW + hw]);
  } else {
    v = hx[(size_t)p * 64 + c];
  }
  a2[(size_t)p * 64 + c] = v;
}

// 5) GRU q + state update h = (1-z)*y + z*tanh(q)
__global__ void k_gru_q(const _Float16* __restrict__ a2,
                        const float* __restrict__ wq, const float* __restrict__ bq,
                        const float* __restrict__ zb, const float* __restrict__ y,
                        float* __restrict__ hb, float* __restrict__ hout) {
  __shared__ __align__(32) _Float16 wqL[18 * 2 * 512];
  __builtin_prefetch(wq, 0, 3);
  stage_gru(wqL, wq);
  __syncthreads();
  int lane = threadIdx.x & 31, wave = threadIdx.x >> 5;
  int pbase = (blockIdx.x * 8 + wave) * 16;
  v8f acc[2] = {};
#pragma unroll 1
  for (int kc = 0; kc < 18; ++kc) {
    v16h a = gA_gru(a2, pbase, kc, lane);
#pragma unroll
    for (int nt = 0; nt < 2; ++nt) acc[nt] = wmma_f16(a, gB_lds(wqL, 2, kc, nt, lane), acc[nt]);
  }
#pragma unroll
  for (int nt = 0; nt < 2; ++nt) {
    int o = nt * 16 + (lane & 15);
    float b0 = bq[o];
#pragma unroll
    for (int v = 0; v < 8; ++v) {
      int m = v + ((lane & 16) ? 8 : 0);
      int p = pbase + m;
      int n = p / HW, hw = p % HW;
      float q = tanhf(acc[nt][v] + b0);
      float z = zb[(size_t)p * 32 + o];
      float yv = y[((size_t)(n * NET + o)) * HW + hw];
      float h = (1.0f - z) * yv + z * q;
      hb[(size_t)p * 32 + o] = h;
      hout[((size_t)(n * NET + o)) * HW + hw] = h;
    }
  }
}

// 6) depthwise 3x3 (32 -> 64) + LayerNorm(64) + exact GELU; wave per pixel
__global__ void k_dw_ln_gelu(const float* __restrict__ hb, const float* __restrict__ dw,
                             const float* __restrict__ db, const float* __restrict__ lg,
                             const float* __restrict__ lb, _Float16* __restrict__ yh) {
  int lane = threadIdx.x & 31, wave = threadIdx.x >> 5;
  int p = blockIdx.x * 8 + wave;
  int n = p / HW, hw = p % HW, h = hw / Ww, w = hw % Ww;
  float vals[2];
#pragma unroll
  for (int s = 0; s < 2; ++s) {
    int oc = lane + s * 32;
    int ic = oc >> 1;
    float acc = db[oc];
#pragma unroll
    for (int t = 0; t < 9; ++t) {
      int hh = h + t / 3 - 1, ww2 = w + t % 3 - 1;
      if (hh >= 0 && hh < Hh && ww2 >= 0 && ww2 < Ww)
        acc += dw[oc * 9 + t] * hb[((size_t)((n * Hh + hh) * Ww + ww2)) * 32 + ic];
    }
    vals[s] = acc;
  }
  float sum = vals[0] + vals[1];
#pragma unroll
  for (int off = 16; off > 0; off >>= 1) sum += __shfl_xor(sum, off, 32);
  float mu = sum * (1.0f / 64.0f);
  float d0 = vals[0] - mu, d1 = vals[1] - mu;
  float vs = d0 * d0 + d1 * d1;
#pragma unroll
  for (int off = 16; off > 0; off >>= 1) vs += __shfl_xor(vs, off, 32);
  float inv = rsqrtf(vs * (1.0f / 64.0f) + LN_EPS);
#pragma unroll
  for (int s = 0; s < 2; ++s) {
    int oc = lane + s * 32;
    float yv = (vals[s] - mu) * inv * lg[oc] + lb[oc];
    yv = 0.5f * yv * (1.0f + erff(yv * 0.70710678118f));
    yh[(size_t)p * 64 + oc] = (_Float16)yv;
  }
}

// 7) input projection x = input_x(NHWC) @ inp_w.T + inp_b  (64->64), f32 out for sampling
__global__ void k_proj(const _Float16* __restrict__ xh, const float* __restrict__ iw,
                       const float* __restrict__ ib, float* __restrict__ xp) {
  __shared__ __align__(32) _Float16 wL[2 * 4 * 512];
  stage_dense(wL, iw, 64, 64, 2, 4);
  __syncthreads();
  int lane = threadIdx.x & 31, wave = threadIdx.x >> 5;
  int pbase = (blockIdx.x * 8 + wave) * 16;
  v8f acc[4] = {};
#pragma unroll
  for (int kc = 0; kc < 2; ++kc) {
    v16h a = gA_nhwc_h(xh, pbase, kc, lane);
#pragma unroll
    for (int nt = 0; nt < 4; ++nt) acc[nt] = wmma_f16(a, gB_lds(wL, 4, kc, nt, lane), acc[nt]);
  }
#pragma unroll
  for (int nt = 0; nt < 4; ++nt) {
    int o = nt * 16 + (lane & 15);
    float b0 = ib[o];
#pragma unroll
    for (int v = 0; v < 8; ++v) {
      int m = v + ((lane & 16) ? 8 : 0);
      xp[(size_t)(pbase + m) * 64 + o] = acc[nt][v] + b0;
    }
  }
}

// 8) offset (64->72) and mask logits (64->36) heads from y
__global__ void k_offmask(const _Float16* __restrict__ yh,
                          const float* __restrict__ ow, const float* __restrict__ ob,
                          const float* __restrict__ mw, const float* __restrict__ mb,
                          float* __restrict__ offb, float* __restrict__ mskb) {
  __shared__ __align__(32) _Float16 oL[2 * 5 * 512];
  __shared__ __align__(32) _Float16 mL[2 * 3 * 512];
  stage_dense(oL, ow, 72, 64, 2, 5);
  stage_dense(mL, mw, 36, 64, 2, 3);
  __syncthreads();
  int lane = threadIdx.x & 31, wave = threadIdx.x >> 5;
  int pbase = (blockIdx.x * 8 + wave) * 16;
  v8f ao[5] = {};
  v8f am[3] = {};
#pragma unroll
  for (int kc = 0; kc < 2; ++kc) {
    v16h a = gA_nhwc_h(yh, pbase, kc, lane);
#pragma unroll
    for (int nt = 0; nt < 5; ++nt) ao[nt] = wmma_f16(a, gB_lds(oL, 5, kc, nt, lane), ao[nt]);
#pragma unroll
    for (int nt = 0; nt < 3; ++nt) am[nt] = wmma_f16(a, gB_lds(mL, 3, kc, nt, lane), am[nt]);
  }
#pragma unroll
  for (int nt = 0; nt < 5; ++nt) {
    int o = nt * 16 + (lane & 15);
    if (o < 72) {
      float b0 = ob[o];
#pragma unroll
      for (int v = 0; v < 8; ++v) {
        int m = v + ((lane & 16) ? 8 : 0);
        offb[(size_t)(pbase + m) * 72 + o] = ao[nt][v] + b0;
      }
    }
  }
#pragma unroll
  for (int nt = 0; nt < 3; ++nt) {
    int o = nt * 16 + (lane & 15);
    if (o < 36) {
      float b0 = mb[o];
#pragma unroll
      for (int v = 0; v < 8; ++v) {
        int m = v + ((lane & 16) ? 8 : 0);
        mskb[(size_t)(pbase + m) * 36 + o] = am[nt][v] + b0;
      }
    }
  }
}

// 9) softmax over the 9 taps of each (pixel, group)
__global__ void k_softmax(float* __restrict__ mskb) {
  int i = blockIdx.x * blockDim.x + threadIdx.x;
  if (i >= NPIX * Gg) return;
  float* m = mskb + (size_t)i * 9;
  float mx = m[0];
#pragma unroll
  for (int t = 1; t < 9; ++t) mx = fmaxf(mx, m[t]);
  float e[9], s = 0.0f;
#pragma unroll
  for (int t = 0; t < 9; ++t) { e[t] = __expf(m[t] - mx); s += e[t]; }
  float inv = 1.0f / s;
#pragma unroll
  for (int t = 0; t < 9; ++t) m[t] = e[t] * inv;
}

// 10) DCNv3 core: bilinear deformable sampling; thread per (pixel, group)
//     output written as f16 NHWC so the final GEMM can b128-load its A tiles
__global__ void k_dcnv3(const float* __restrict__ xp, const float* __restrict__ offb,
                        const float* __restrict__ mskb, _Float16* __restrict__ valh) {
  int i = blockIdx.x * blockDim.x + threadIdx.x;
  if (i >= NPIX * Gg) return;
  int p = i >> 2, g = i & 3;
  int n = p / HW, hw = p % HW, h = hw / Ww, w = hw % Ww;
  const float* off = offb + (size_t)p * 72 + g * 18;
  const float* msk = mskb + (size_t)p * 36 + g * 9;
  float acc[GC] = {};
#pragma unroll 1
  for (int t = 0; t < 9; ++t) {
    // tap order is w-major (torch meshgrid 'ij'): dx = t/3-1, dy = t%3-1
    float dx = (float)(t / 3) - 1.0f, dy = (float)(t % 3) - 1.0f;
    // padded pixel coords: px = loc_x*Win - 0.5 = w + dx + 1 + raw_offset_x
    float px = (float)w + dx + 1.0f + off[t * 2 + 0];
    float py = (float)h + dy + 1.0f + off[t * 2 + 1];
    float fx = floorf(px), fy = floorf(py);
    int ix0 = (int)fx, iy0 = (int)fy;
    float wx1 = px - fx, wy1 = py - fy, wx0 = 1.0f - wx1, wy0 = 1.0f - wy1;
    float mv = msk[t];
#pragma unroll
    for (int sy = 0; sy < 2; ++sy) {
#pragma unroll
      for (int sx = 0; sx < 2; ++sx) {
        int ix = ix0 + sx - 1;   // unpad (pad ring and OOB both read as zero)
        int iy = iy0 + sy - 1;
        if (ix >= 0 && ix < Ww && iy >= 0 && iy < Hh) {
          float wgt = (sx ? wx1 : wx0) * (sy ? wy1 : wy0) * mv;
          const float* src = xp + ((size_t)((n * Hh + iy) * Ww + ix)) * 64 + g * GC;
#pragma unroll
          for (int c = 0; c < GC; ++c) acc[c] += wgt * src[c];
        }
      }
    }
  }
  _Float16* dst = valh + (size_t)p * 64 + g * GC;
#pragma unroll
  for (int c = 0; c < GC; ++c) dst[c] = (_Float16)acc[c];
}

// 11) output projection (64->64) + SiLU, store NCHW to d_out
__global__ void k_out(const _Float16* __restrict__ valh, const float* __restrict__ ow,
                      const float* __restrict__ ob, float* __restrict__ out) {
  __shared__ __align__(32) _Float16 wL[2 * 4 * 512];
  stage_dense(wL, ow, 64, 64, 2, 4);
  __syncthreads();
  int lane = threadIdx.x & 31, wave = threadIdx.x >> 5;
  int pbase = (blockIdx.x * 8 + wave) * 16;
  v8f acc[4] = {};
#pragma unroll
  for (int kc = 0; kc < 2; ++kc) {
    v16h a = gA_nhwc_h(valh, pbase, kc, lane);
#pragma unroll
    for (int nt = 0; nt < 4; ++nt) acc[nt] = wmma_f16(a, gB_lds(wL, 4, kc, nt, lane), acc[nt]);
  }
#pragma unroll
  for (int nt = 0; nt < 4; ++nt) {
    int o = nt * 16 + (lane & 15);
    float b0 = ob[o];
#pragma unroll
    for (int v = 0; v < 8; ++v) {
      int m = v + ((lane & 16) ? 8 : 0);
      int p = pbase + m;
      int n = p / HW, hw = p % HW;
      float x = acc[nt][v] + b0;
      out[((size_t)(n * Cc + o)) * HW + hw] = x * sigm(x);
    }
  }
}

// =====================================================================
extern "C" void kernel_launch(void* const* d_in, const int* in_sizes, int n_in,
                              void* d_out, int out_size, void* d_ws, size_t ws_size,
                              hipStream_t stream) {
  const float* input_x = (const float*)d_in[0];
  const float* input_y = (const float*)d_in[1];
  const float* enc_w  = (const float*)d_in[2];
  const float* enc_b  = (const float*)d_in[3];
  const float* bn_g   = (const float*)d_in[4];
  const float* bn_b   = (const float*)d_in[5];
  const float* bn_m   = (const float*)d_in[6];
  const float* bn_v   = (const float*)d_in[7];
  const float* gru_wz = (const float*)d_in[8];
  const float* gru_bz = (const float*)d_in[9];
  const float* gru_wr = (const float*)d_in[10];
  const float* gru_br = (const float*)d_in[11];
  const float* gru_wq = (const float*)d_in[12];
  const float* gru_bq = (const float*)d_in[13];
  const float* dw_w   = (const float*)d_in[14];
  const float* dw_b   = (const float*)d_in[15];
  const float* ln_g   = (const float*)d_in[16];
  const float* ln_b   = (const float*)d_in[17];
  const float* off_w  = (const float*)d_in[18];
  const float* off_b  = (const float*)d_in[19];
  const float* mask_w = (const float*)d_in[20];
  const float* mask_b = (const float*)d_in[21];
  const float* inp_w  = (const float*)d_in[22];
  const float* inp_b  = (const float*)d_in[23];
  const float* out_w  = (const float*)d_in[24];
  const float* out_b  = (const float*)d_in[25];
  (void)in_sizes; (void)n_in; (void)out_size; (void)ws_size;

  // workspace layout (all intermediates stay L2-resident, ~107 MB)
  char* ws = (char*)d_ws;
  _Float16* hx   = (_Float16*)(ws + (size_t)0);           //  8 MB  hx = [y | x_en] f16 NHWC64
  _Float16* xh   = (_Float16*)(ws + ((size_t)8  << 20));  //  8 MB  input_x f16 NHWC64
  float*    zb   = (float*)(ws + ((size_t)16 << 20));     //  8 MB
  float*    rb   = (float*)(ws + ((size_t)24 << 20));     //  8 MB
  _Float16* a2   = (_Float16*)(ws + ((size_t)32 << 20));  //  8 MB
  float*    hb   = (float*)(ws + ((size_t)40 << 20));     //  8 MB  h NHWC32
  _Float16* yh   = (_Float16*)(ws + ((size_t)48 << 20));  //  8 MB  y f16 NHWC64
  float*    xp   = (float*)(ws + ((size_t)56 << 20));     // 16 MB  xproj f32 NHWC64
  float*    offb = (float*)(ws + ((size_t)72 << 20));     // 18 MB  NHWC72
  float*    mskb = (float*)(ws + ((size_t)90 << 20));     //  9 MB  NHWC36
  _Float16* valh = (_Float16*)(ws + ((size_t)99 << 20));  //  8 MB  f16 NHWC64

  float* out0 = (float*)d_out;                            // SiLU output, NCHW 4x64x128x128
  float* hout = out0 + (size_t)Nn * Cc * HW;              // h,          NCHW 4x32x128x128

  const int BLK = 256;
  const int GEMM_BLOCKS = NPIX / 128;                     // 8 waves * 16 px each

  k_pack_xy<<<(NPIX * 64 + BLK - 1) / BLK, BLK, 0, stream>>>(input_x, input_y, xh, hx);
  k_encoder<<<GEMM_BLOCKS, BLK, 0, stream>>>(xh, enc_w, enc_b, bn_g, bn_b, bn_m, bn_v, hx);
  k_gru_zr<<<GEMM_BLOCKS, BLK, 0, stream>>>(hx, gru_wz, gru_bz, gru_wr, gru_br, zb, rb);
  k_build_a2<<<(NPIX * 64 + BLK - 1) / BLK, BLK, 0, stream>>>(input_y, rb, hx, a2);
  k_gru_q<<<GEMM_BLOCKS, BLK, 0, stream>>>(a2, gru_wq, gru_bq, zb, input_y, hb, hout);
  k_dw_ln_gelu<<<NPIX / 8, BLK, 0, stream>>>(hb, dw_w, dw_b, ln_g, ln_b, yh);
  k_proj<<<GEMM_BLOCKS, BLK, 0, stream>>>(xh, inp_w, inp_b, xp);
  k_offmask<<<GEMM_BLOCKS, BLK, 0, stream>>>(yh, off_w, off_b, mask_w, mask_b, offb, mskb);
  k_softmax<<<(NPIX * Gg + BLK - 1) / BLK, BLK, 0, stream>>>(mskb);
  k_dcnv3<<<(NPIX * Gg + BLK - 1) / BLK, BLK, 0, stream>>>(xp, offb, mskb, valh);
  k_out<<<GEMM_BLOCKS, BLK, 0, stream>>>(valh, out_w, out_b, out0);
}
